// SelfAttention_42099269436054
// MI455X (gfx1250) — compile-verified, matrix operand order checked by reading, not run on hardware
//
#include <hip/hip_runtime.h>
#include <hip/hip_bf16.h>

// Self-attention (B=8, C=512, 32x32 spatial, 8 heads, hd=64) for gfx1250.
// All matmuls on v_wmma_f32_16x16x32_f16; all tile staging through the
// Tensor Data Mover (tensor_load_to_lds), double-buffered in LDS so the DMA
// overlaps WMMA work; completion via s_wait_tensorcnt + workgroup barrier.

#define B_   8
#define C_   512
#define S_   1024      // 32*32
#define NH_  8
#define HD_  64
#define C3_  1536

typedef __attribute__((ext_vector_type(16))) _Float16 v16h;
typedef __attribute__((ext_vector_type(8)))  _Float16 v8h;
typedef __attribute__((ext_vector_type(8)))  float    v8f;
typedef __attribute__((ext_vector_type(4)))  unsigned int u32x4;
typedef __attribute__((ext_vector_type(8)))  unsigned int u32x8;

union Frag16 { v16h v; v8h h[2]; _Float16 e[16]; };

__device__ __forceinline__ v8f vzero8() {
  v8f z; for (int i = 0; i < 8; ++i) z[i] = 0.0f; return z;
}

// 16x32 f16 fragment from row-major storage (ld in elements), ISA 7.12.2
// layout: lane L holds row (L&15); lanes 0-15 carry K 0-7 / 16-23, lanes
// 16-31 carry K 8-15 / 24-31. Two 16B loads per lane.
__device__ __forceinline__ v16h load_frag(const _Float16* __restrict__ base,
                                          int ld, int row0, int col0, int lane) {
  int r  = lane & 15;
  int hf = lane >> 4;
  const _Float16* p = base + (size_t)(row0 + r) * ld + col0 + hf * 8;
  Frag16 f;
  f.h[0] = *(const v8h*)(p);
  f.h[1] = *(const v8h*)(p + 16);
  return f.v;
}

__device__ __forceinline__ v8f wmma_f16(v16h a, v16h b, v8f c) {
  return __builtin_amdgcn_wmma_f32_16x16x32_f16(false, a, false, b, (short)0, c,
                                                false, false);
}

// ---------------- Tensor Data Mover: 2D f16 tile -> LDS --------------------
// D# per CDNA5 ISA; data_size=1 (2B elements).
// pad_code: pad after 2^pad_code * 8 bytes; pad_amount fixed 8 DWORDs (32B),
// so a 64-elem (128B) row with pad_code=4 gives LDS row stride 160B = 80 f16.
#define LDS_LD64 80   // padded LDS row stride (f16) for 64-element tile rows

__device__ __forceinline__ void tdm_load_2d(unsigned lds_off,
                                            const _Float16* gptr,
                                            unsigned tile_d0, unsigned tile_d1,
                                            unsigned pad_code,
                                            unsigned tensor_d0, unsigned tensor_d1,
                                            unsigned stride0) {
  unsigned long long ga = (unsigned long long)(const void*)gptr;
  u32x4 g0;
  g0[0] = 1u;                                   // count=1, user mode
  g0[1] = lds_off;                              // lds_addr (bytes)
  g0[2] = (unsigned)ga;                         // global_addr[31:0]
  g0[3] = (unsigned)(ga >> 32) | (2u << 30);    // global_addr[56:32] | type=2
  u32x8 g1;
  g1[0] = (1u << 16)               // data_size = 1 (2 bytes)
        | (1u << 20)               // pad_enable
        | (pad_code << 22)         // pad_interval
        | (7u << 25);              // pad_amount: 8 DWORDs
  g1[1] = (tensor_d0 & 0xFFFFu) << 16;
  g1[2] = (tensor_d0 >> 16) | ((tensor_d1 & 0xFFFFu) << 16);
  g1[3] = (tensor_d1 >> 16) | (tile_d0 << 16);
  g1[4] = tile_d1 & 0xFFFFu;                    // tile_dim1 (tile_dim2=0)
  g1[5] = stride0;
  g1[6] = 0u;
  g1[7] = 0u;
  asm volatile("tensor_load_to_lds %0, %1" :: "s"(g0), "s"(g1) : "memory");
}

__device__ __forceinline__ unsigned lds_offset_of(const void* p) {
  return (unsigned)(unsigned long long)p;   // low 32 bits = LDS byte offset
}

// ---------------- Stage 0: layout conversion -------------------------------

__global__ void cvt_x_kernel(const float* __restrict__ x, _Float16* __restrict__ xs) {
  __shared__ float tile[32][33];
  int b  = blockIdx.z;
  int c0 = blockIdx.y * 32;
  int s0 = blockIdx.x * 32;
  int tx = threadIdx.x, ty = threadIdx.y;
  const float* xp = x + ((size_t)b * C_ + c0) * S_ + s0;
  for (int i = 0; i < 4; ++i) {
    int r = ty * 4 + i;
    tile[r][tx] = xp[(size_t)r * S_ + tx];
  }
  __syncthreads();
  _Float16* xsp = xs + ((size_t)b * S_ + s0) * C_ + c0;
  for (int i = 0; i < 4; ++i) {
    int r = ty * 4 + i;
    xsp[(size_t)r * C_ + tx] = (_Float16)tile[tx][r];
  }
}

__global__ void cvt_f16_kernel(const float* __restrict__ src,
                               _Float16* __restrict__ dst, int n) {
  int i = blockIdx.x * blockDim.x + threadIdx.x;
  if (i < n) dst[i] = (_Float16)src[i];
}

// ---------------- Stage 1: QKV projection ----------------------------------
// WG tile 128(j) x 64(s), 8 waves in 4x2, wave tile 32x32 (2x2 frags).
// K-step 64; A/B tiles double-buffered in LDS via TDM (DMA overlaps WMMA).
__global__ void __launch_bounds__(256)
qkv_gemm_kernel(const _Float16* __restrict__ xs,
                const _Float16* __restrict__ w,
                const float* __restrict__ bias,
                _Float16* __restrict__ q,
                _Float16* __restrict__ k,
                _Float16* __restrict__ vT) {
  __shared__ __align__(16) _Float16 As[2][128 * LDS_LD64];   // 40 KB
  __shared__ __align__(16) _Float16 Bs[2][64 * LDS_LD64];    // 20 KB
  int tid = threadIdx.x;
  int waveId = tid >> 5, lane = tid & 31;
  int mB = blockIdx.x % 12;                 // j block (128)
  int nB = (blockIdx.x / 12) % 16;          // s block (64)
  int b  = blockIdx.x / (12 * 16);
  int j0 = mB * 128, s0 = nB * 64;
  const _Float16* xb = xs + (size_t)b * S_ * C_;
  const _Float16* wj = w + (size_t)j0 * C_;
  int mw = (waveId & 3) * 32;
  int nw = (waveId >> 2) * 32;
  unsigned aLds[2] = { lds_offset_of(&As[0][0]), lds_offset_of(&As[1][0]) };
  unsigned bLds[2] = { lds_offset_of(&Bs[0][0]), lds_offset_of(&Bs[1][0]) };

  if (waveId == 0) {
    tdm_load_2d(aLds[0], wj,                   64, 128, 4, C_, C3_, C_);
    tdm_load_2d(bLds[0], xb + (size_t)s0 * C_, 64, 64,  4, C_, S_,  C_);
    __builtin_amdgcn_s_wait_tensorcnt(0);
  }
  __syncthreads();

  v8f acc00 = vzero8(), acc01 = vzero8(), acc10 = vzero8(), acc11 = vzero8();
  for (int step = 0; step < 8; ++step) {
    int cur = step & 1, nxt = cur ^ 1;
    int k0 = step * 64;
    if (waveId == 0 && step < 7) {
      tdm_load_2d(aLds[nxt], wj + k0 + 64,                   64, 128, 4, C_, C3_, C_);
      tdm_load_2d(bLds[nxt], xb + (size_t)s0 * C_ + k0 + 64, 64, 64,  4, C_, S_,  C_);
    }
    for (int kc = 0; kc < 2; ++kc) {
      v16h a0 = load_frag(As[cur], LDS_LD64, mw,      kc * 32, lane);
      v16h a1 = load_frag(As[cur], LDS_LD64, mw + 16, kc * 32, lane);
      v16h b0 = load_frag(Bs[cur], LDS_LD64, nw,      kc * 32, lane);
      v16h b1 = load_frag(Bs[cur], LDS_LD64, nw + 16, kc * 32, lane);
      acc00 = wmma_f16(a0, b0, acc00);
      acc01 = wmma_f16(a0, b1, acc01);
      acc10 = wmma_f16(a1, b0, acc10);
      acc11 = wmma_f16(a1, b1, acc11);
    }
    if (waveId == 0) __builtin_amdgcn_s_wait_tensorcnt(0);
    __syncthreads();
  }

  int n  = lane & 15;
  int hf = lane >> 4;
  v8f accs[2][2] = { { acc00, acc01 }, { acc10, acc11 } };
  for (int fm = 0; fm < 2; ++fm) {
    int jbase = j0 + mw + fm * 16;          // stays within one part & head
    int part  = jbase >> 9;
    int hh    = (jbase & 511) >> 6;
    int dbase = (jbase & 63) + hf * 8;
    for (int fn = 0; fn < 2; ++fn) {
      int s = s0 + nw + fn * 16 + n;
      v8f acc = accs[fm][fn];
      for (int r = 0; r < 8; ++r) {
        int j = jbase + hf * 8 + r;
        float val = acc[r] + bias[j];
        int d = dbase + r;
        if (part == 0)
          q[(((size_t)b * NH_ + hh) * S_ + s) * HD_ + d] = (_Float16)val;
        else if (part == 1)
          k[(((size_t)b * NH_ + hh) * S_ + s) * HD_ + d] = (_Float16)val;
        else
          vT[(((size_t)b * NH_ + hh) * HD_ + d) * S_ + s] = (_Float16)val;
      }
    }
  }
}

// ---------------- Stage 2: flash attention ---------------------------------
// Block = 256 queries of one (b,h): 8 waves x 32 queries. K/V tiles (64 keys)
// staged in LDS by TDM, double-buffered, shared by all 8 waves (8x traffic
// cut). S^T = K*Q^T keeps softmax in-lane (+ one shfl_xor(16)); the S^T
// D-frag row split matches the B-frag K split so P^T repacks in-lane.
__global__ void __launch_bounds__(256)
attn_kernel(const _Float16* __restrict__ q,
            const _Float16* __restrict__ k,
            const _Float16* __restrict__ vT,
            _Float16* __restrict__ av) {
  __shared__ __align__(16) _Float16 Ks[2][64 * LDS_LD64];    // 20 KB
  __shared__ __align__(16) _Float16 Vs[2][64 * LDS_LD64];    // 20 KB
  int tid = threadIdx.x;
  int waveId = tid >> 5, lane = tid & 31;
  int qB = blockIdx.x & 3;                  // 256-query block
  int h  = (blockIdx.x >> 2) & 7;
  int b  = blockIdx.x >> 5;
  int q0 = qB * 256 + waveId * 32;

  const _Float16* qb = q  + ((size_t)b * NH_ + h) * S_  * HD_;
  const _Float16* kb = k  + ((size_t)b * NH_ + h) * S_  * HD_;
  const _Float16* vb = vT + ((size_t)b * NH_ + h) * HD_ * S_;
  unsigned kLds[2] = { lds_offset_of(&Ks[0][0]), lds_offset_of(&Ks[1][0]) };
  unsigned vLds[2] = { lds_offset_of(&Vs[0][0]), lds_offset_of(&Vs[1][0]) };

  v16h qf[2][2];
  for (int j = 0; j < 2; ++j) {
    qf[j][0] = load_frag(qb, HD_, q0 + j * 16, 0,  lane);
    qf[j][1] = load_frag(qb, HD_, q0 + j * 16, 32, lane);
  }

  v8f o[4][2];
  for (int dt = 0; dt < 4; ++dt) for (int j = 0; j < 2; ++j) o[dt][j] = vzero8();
  float mrun[2] = { -1e30f, -1e30f };
  float lrun[2] = { 0.0f, 0.0f };
  const float scale = 0.125f;   // 1/sqrt(64)

  if (waveId == 0) {
    tdm_load_2d(kLds[0], kb, 64, 64, 4, HD_, S_, HD_);  // 64 keys x 64 d
    tdm_load_2d(vLds[0], vb, 64, 64, 4, S_, HD_, S_);   // 64 d x 64 keys
    __builtin_amdgcn_s_wait_tensorcnt(0);
  }
  __syncthreads();

  for (int step = 0; step < 16; ++step) {               // 64 keys per step
    int cur = step & 1, nxt = cur ^ 1;
    int key0 = step * 64;
    if (waveId == 0 && step < 15) {
      tdm_load_2d(kLds[nxt], kb + (size_t)(key0 + 64) * HD_, 64, 64, 4, HD_, S_, HD_);
      tdm_load_2d(vLds[nxt], vb + (key0 + 64),               64, 64, 4, S_, HD_, S_);
    }
    for (int g = 0; g < 2; ++g) {                       // two 32-key groups
      int kl = g * 32;
      v16h a00 = load_frag(Ks[cur], LDS_LD64, kl,      0,  lane);
      v16h a01 = load_frag(Ks[cur], LDS_LD64, kl,      32, lane);
      v16h a10 = load_frag(Ks[cur], LDS_LD64, kl + 16, 0,  lane);
      v16h a11 = load_frag(Ks[cur], LDS_LD64, kl + 16, 32, lane);

      Frag16 pb[2];
      for (int j = 0; j < 2; ++j) {
        v8f st0 = vzero8(), st1 = vzero8();
        st0 = wmma_f16(a00, qf[j][0], st0);
        st0 = wmma_f16(a01, qf[j][1], st0);
        st1 = wmma_f16(a10, qf[j][0], st1);
        st1 = wmma_f16(a11, qf[j][1], st1);

        float mloc = -1e30f;
        for (int r = 0; r < 8; ++r) {
          st0[r] *= scale; st1[r] *= scale;
          mloc = fmaxf(mloc, fmaxf(st0[r], st1[r]));
        }
        mloc = fmaxf(mloc, __shfl_xor(mloc, 16, 32));
        float mnew  = fmaxf(mrun[j], mloc);
        float alpha = __expf(mrun[j] - mnew);

        float lloc = 0.0f;
        for (int r = 0; r < 8; ++r) {
          float p0 = __expf(st0[r] - mnew);
          float p1 = __expf(st1[r] - mnew);
          lloc += p0 + p1;
          pb[j].e[r]     = (_Float16)p0;
          pb[j].e[8 + r] = (_Float16)p1;
        }
        lloc += __shfl_xor(lloc, 16, 32);
        lrun[j] = lrun[j] * alpha + lloc;
        mrun[j] = mnew;
        for (int dt = 0; dt < 4; ++dt)
          for (int r = 0; r < 8; ++r) o[dt][j][r] *= alpha;
      }

      v16h vf0 = load_frag(Vs[cur], LDS_LD64, 0,  kl, lane);
      v16h vf1 = load_frag(Vs[cur], LDS_LD64, 16, kl, lane);
      v16h vf2 = load_frag(Vs[cur], LDS_LD64, 32, kl, lane);
      v16h vf3 = load_frag(Vs[cur], LDS_LD64, 48, kl, lane);
      for (int j = 0; j < 2; ++j) {
        o[0][j] = wmma_f16(vf0, pb[j].v, o[0][j]);
        o[1][j] = wmma_f16(vf1, pb[j].v, o[1][j]);
        o[2][j] = wmma_f16(vf2, pb[j].v, o[2][j]);
        o[3][j] = wmma_f16(vf3, pb[j].v, o[3][j]);
      }
    }
    if (waveId == 0) __builtin_amdgcn_s_wait_tensorcnt(0);
    __syncthreads();
  }

  int n  = lane & 15;
  int hf = lane >> 4;
  for (int j = 0; j < 2; ++j) {
    float inv = 1.0f / lrun[j];
    _Float16* avp = av + ((size_t)b * S_ + q0 + j * 16 + n) * C_ + h * HD_ + hf * 8;
    for (int dt = 0; dt < 4; ++dt) {
      v8h pack;
      for (int r = 0; r < 8; ++r) pack[r] = (_Float16)(o[dt][j][r] * inv);
      *(v8h*)(avp + dt * 16) = pack;
    }
  }
}

// ---------------- Stage 3: output projection -------------------------------
// out (B,C,S) f32 in transposed orientation -> coalesced f32 stores.
// Same double-buffered TDM GEMM as stage 1.
__global__ void __launch_bounds__(256)
out_gemm_kernel(const _Float16* __restrict__ av,
                const _Float16* __restrict__ wo,
                const float* __restrict__ bo,
                float* __restrict__ out) {
  __shared__ __align__(16) _Float16 As[2][128 * LDS_LD64];
  __shared__ __align__(16) _Float16 Bs[2][64 * LDS_LD64];
  int tid = threadIdx.x;
  int waveId = tid >> 5, lane = tid & 31;
  int mB = blockIdx.x % 4;                  // c block (128)
  int nB = (blockIdx.x / 4) % 16;           // s block (64)
  int b  = blockIdx.x / (4 * 16);
  int c0 = mB * 128, s0 = nB * 64;
  const _Float16* avb = av + (size_t)b * S_ * C_;
  const _Float16* wc  = wo + (size_t)c0 * C_;
  int mw = (waveId & 3) * 32;
  int nw = (waveId >> 2) * 32;
  unsigned aLds[2] = { lds_offset_of(&As[0][0]), lds_offset_of(&As[1][0]) };
  unsigned bLds[2] = { lds_offset_of(&Bs[0][0]), lds_offset_of(&Bs[1][0]) };

  if (waveId == 0) {
    tdm_load_2d(aLds[0], wc,                    64, 128, 4, C_, C_, C_);
    tdm_load_2d(bLds[0], avb + (size_t)s0 * C_, 64, 64,  4, C_, S_, C_);
    __builtin_amdgcn_s_wait_tensorcnt(0);
  }
  __syncthreads();

  v8f acc00 = vzero8(), acc01 = vzero8(), acc10 = vzero8(), acc11 = vzero8();
  for (int step = 0; step < 8; ++step) {
    int cur = step & 1, nxt = cur ^ 1;
    int k0 = step * 64;
    if (waveId == 0 && step < 7) {
      tdm_load_2d(aLds[nxt], wc  + k0 + 64,                   64, 128, 4, C_, C_, C_);
      tdm_load_2d(bLds[nxt], avb + (size_t)s0 * C_ + k0 + 64, 64, 64,  4, C_, S_, C_);
    }
    for (int kc = 0; kc < 2; ++kc) {
      v16h a0 = load_frag(As[cur], LDS_LD64, mw,      kc * 32, lane);
      v16h a1 = load_frag(As[cur], LDS_LD64, mw + 16, kc * 32, lane);
      v16h b0 = load_frag(Bs[cur], LDS_LD64, nw,      kc * 32, lane);
      v16h b1 = load_frag(Bs[cur], LDS_LD64, nw + 16, kc * 32, lane);
      acc00 = wmma_f16(a0, b0, acc00);
      acc01 = wmma_f16(a0, b1, acc01);
      acc10 = wmma_f16(a1, b0, acc10);
      acc11 = wmma_f16(a1, b1, acc11);
    }
    if (waveId == 0) __builtin_amdgcn_s_wait_tensorcnt(0);
    __syncthreads();
  }

  int n  = lane & 15;
  int hf = lane >> 4;
  v8f accs[2][2] = { { acc00, acc01 }, { acc10, acc11 } };
  for (int fm = 0; fm < 2; ++fm) {
    for (int fn = 0; fn < 2; ++fn) {
      v8f acc = accs[fm][fn];
      for (int r = 0; r < 8; ++r) {
        int c = c0 + mw + fm * 16 + hf * 8 + r;
        out[((size_t)b * C_ + c) * S_ + s0 + nw + fn * 16 + n] = acc[r] + bo[c];
      }
    }
  }
}

// ---------------- Launch ---------------------------------------------------

extern "C" void kernel_launch(void* const* d_in, const int* in_sizes, int n_in,
                              void* d_out, int out_size, void* d_ws, size_t ws_size,
                              hipStream_t stream) {
  (void)in_sizes; (void)n_in; (void)out_size; (void)ws_size;
  const float* x      = (const float*)d_in[0];
  const float* w_qkv  = (const float*)d_in[1];
  const float* b_qkv  = (const float*)d_in[2];
  const float* w_o    = (const float*)d_in[3];
  const float* b_o    = (const float*)d_in[4];
  float* out = (float*)d_out;

  char* ws = (char*)d_ws;
  size_t off = 0;
  _Float16* xs   = (_Float16*)(ws + off); off += (size_t)B_ * S_ * C_ * 2;
  _Float16* wq16 = (_Float16*)(ws + off); off += (size_t)C3_ * C_ * 2;
  _Float16* wo16 = (_Float16*)(ws + off); off += (size_t)C_ * C_ * 2;
  _Float16* qh   = (_Float16*)(ws + off); off += (size_t)B_ * NH_ * S_ * HD_ * 2;
  _Float16* kh   = (_Float16*)(ws + off); off += (size_t)B_ * NH_ * S_ * HD_ * 2;
  _Float16* vTh  = (_Float16*)(ws + off); off += (size_t)B_ * NH_ * HD_ * S_ * 2;
  _Float16* av   = (_Float16*)(ws + off); off += (size_t)B_ * S_ * C_ * 2;

  cvt_x_kernel<<<dim3(S_ / 32, C_ / 32, B_), dim3(32, 8), 0, stream>>>(x, xs);
  cvt_f16_kernel<<<(C3_ * C_ + 255) / 256, 256, 0, stream>>>(w_qkv, wq16, C3_ * C_);
  cvt_f16_kernel<<<(C_ * C_ + 255) / 256, 256, 0, stream>>>(w_o, wo16, C_ * C_);

  // Stage 1: 12 (j/128) * 16 (s/64) * 8 (b) = 1536 blocks
  qkv_gemm_kernel<<<1536, 256, 0, stream>>>(xs, wq16, b_qkv, qh, kh, vTh);

  // Stage 2: 4 (q/256) * 8 (h) * 8 (b) = 256 blocks
  attn_kernel<<<256, 256, 0, stream>>>(qh, kh, vTh, av);

  // Stage 3: 4 (c/128) * 16 (s/64) * 8 (b) = 512 blocks
  out_gemm_kernel<<<512, 256, 0, stream>>>(av, wo16, b_o, out);
}
